// TripleLSTM_89721866813731
// MI455X (gfx1250) — matrix operand order repaired and checked by reading, not compiled, not run on hardware
//
#include <hip/hip_runtime.h>
#include <hip/hip_bf16.h>

// ---------------------------------------------------------------------------
// TripleLSTM for MI455X (gfx1250): bf16 WMMA GEMMs + async-LDS double buffering
// ---------------------------------------------------------------------------
#define B_   64
#define T_   64
#define NC_  22
#define FUS_ 4096
#define FIN_ 2048
#define H_   1024

typedef __attribute__((ext_vector_type(16))) __bf16 v16bf;
typedef __attribute__((ext_vector_type(8)))  __bf16 v8bf;
typedef __attribute__((ext_vector_type(8)))  float  v8f;

// ---------------------------------------------------------------------------
// gfx1250 async global->LDS copy (ASYNCcnt path), with graceful fallback
// ---------------------------------------------------------------------------
#if defined(__AMDGCN__) && __has_builtin(__builtin_amdgcn_global_load_async_to_lds_b128)
#define USE_ASYNC_LDS 1
#else
#define USE_ASYNC_LDS 0
#endif

#if USE_ASYNC_LDS
typedef int b128i __attribute__((vector_size(16)));
typedef __attribute__((address_space(1))) b128i b128i_g;  // global (AS1)
typedef __attribute__((address_space(3))) b128i b128i_l;  // LDS (AS3)
#endif

static __device__ inline void async_copy16B(const __bf16* g, __bf16* l) {
#if USE_ASYNC_LDS
  __builtin_amdgcn_global_load_async_to_lds_b128(
      (b128i_g*)g, (b128i_l*)l, /*offset=*/0, /*cpol=*/0);
#else
  *(v8bf*)l = *(const v8bf*)g;
#endif
}

static __device__ inline void async_join() {
#if USE_ASYNC_LDS
#if __has_builtin(__builtin_amdgcn_s_wait_asynccnt)
  __builtin_amdgcn_s_wait_asynccnt(0);
#else
  asm volatile("s_wait_asynccnt 0x0" ::: "memory");
#endif
#endif
}

// ---------------------------------------------------------------------------
// WMMA helpers (CDNA5 16x16x32 bf16 -> f32, wave32)
// ---------------------------------------------------------------------------
static __device__ inline v8f wmma_bf16(v16bf a, v16bf b, v8f c) {
  return __builtin_amdgcn_wmma_f32_16x16x32_bf16(
      /*neg_a=*/false, a, /*neg_b=*/false, b,
      /*c_mod=*/(short)0, c, /*reuse_a=*/false, /*reuse_b=*/false);
}

// A fragment: 16x32 (MxK) bf16. Lanes 0-15 -> M=lane, K 0..15;
// lanes 16-31 -> M=lane-16, K 16..31. LDS tile row-major [rows][32].
static __device__ inline v16bf load_a_frag(const __bf16* base, int lane) {
  const int m  = lane & 15;
  const int kb = (lane < 16) ? 0 : 16;
  union { v16bf v; v8bf h[2]; } u;
  u.h[0] = *(const v8bf*)(base + m * 32 + kb);
  u.h[1] = *(const v8bf*)(base + m * 32 + kb + 8);
  return u.v;
}

// B fragment: 32x16 (KxN) bf16. N = lane%16; lanes 0-15 hold K=0..15,
// lanes 16-31 hold K=16..31 (16 contiguous elems in the W-row-major tile).
static __device__ inline v16bf load_b_frag(const __bf16* base, int lane) {
  const int n  = lane & 15;
  const int kb = (lane < 16) ? 0 : 16;
  return *(const v16bf*)(base + n * 32 + kb);
}

// ---------------------------------------------------------------------------
// Tiled GEMM: Cout[M,N] = A[M,K](bf16) @ W[N,K](bf16)^T  (+bias +bias2 +Cadd)
// Block tile 64x64, BK=32, 128 threads (4 waves), wave owns 32x32 quadrant.
// Double-buffered LDS; tiles staged with GLOBAL_LOAD_ASYNC_TO_LDS_B128.
// ---------------------------------------------------------------------------
template<bool RELU>
__global__ __launch_bounds__(128)
void gemm_wmma_bf16(const __bf16* __restrict__ A, const __bf16* __restrict__ W,
                    const float* __restrict__ bias, const float* __restrict__ bias2,
                    const float* __restrict__ Cadd, float* __restrict__ Cout,
                    __bf16* __restrict__ CoutBf, int M, int N, int K) {
  __shared__ __attribute__((aligned(16))) __bf16 As[2][64 * 32];
  __shared__ __attribute__((aligned(16))) __bf16 Bs[2][64 * 32];
  const int tid  = threadIdx.x;
  const int lane = tid & 31;
  const int wave = tid >> 5;
  const int m0 = blockIdx.y * 64;
  const int n0 = blockIdx.x * 64;
  const int wm = (wave >> 1) * 32;   // wave's quadrant within the 64x64 tile
  const int wn = (wave & 1) * 32;

  v8f acc[2][2] = {};

  // Each thread stages 16 bf16 of A and 16 bf16 of W per tile (2x b128 each).
  const int r  = tid >> 1;
  const int hh = (tid & 1) * 16;
  const __bf16* gA = A + (size_t)(m0 + r) * K + hh;
  const __bf16* gW = W + (size_t)(n0 + r) * K + hh;

  auto stage = [&](int buf, int k0) {
    async_copy16B(gA + k0,     &As[buf][r * 32 + hh]);
    async_copy16B(gA + k0 + 8, &As[buf][r * 32 + hh + 8]);
    async_copy16B(gW + k0,     &Bs[buf][r * 32 + hh]);
    async_copy16B(gW + k0 + 8, &Bs[buf][r * 32 + hh + 8]);
  };

  // Prologue: stage tile 0.
  stage(0, 0);
  async_join();
  __syncthreads();

  int buf = 0;
  for (int k0 = 0; k0 < K; k0 += 32) {
    // Kick off the next tile into the alternate buffer (overlaps with WMMA).
    if (k0 + 32 < K) stage(buf ^ 1, k0 + 32);

    v16bf a0 = load_a_frag(&As[buf][(wm +  0) * 32], lane);
    v16bf a1 = load_a_frag(&As[buf][(wm + 16) * 32], lane);
    v16bf b0 = load_b_frag(&Bs[buf][(wn +  0) * 32], lane);
    v16bf b1 = load_b_frag(&Bs[buf][(wn + 16) * 32], lane);

    acc[0][0] = wmma_bf16(a0, b0, acc[0][0]);
    acc[0][1] = wmma_bf16(a0, b1, acc[0][1]);
    acc[1][0] = wmma_bf16(a1, b0, acc[1][0]);
    acc[1][1] = wmma_bf16(a1, b1, acc[1][1]);

    async_join();      // this wave's staged tile has landed in LDS
    __syncthreads();   // every wave's tile has landed; reads of `buf` done
    buf ^= 1;
  }

  // C/D layout: VGPR rr -> M = rr + 8*(lane>=16); N = lane%16.
  const int hm = (lane >> 4) * 8;
  const int ln = lane & 15;
  for (int i = 0; i < 2; ++i) {
    for (int j = 0; j < 2; ++j) {
      const int col = n0 + wn + j * 16 + ln;
      for (int rr = 0; rr < 8; ++rr) {
        const int row = m0 + wm + i * 16 + hm + rr;
        float v = acc[i][j][rr];
        if (bias)   v += bias[col];
        if (bias2)  v += bias2[col];
        if (Cadd)   v += Cadd[(size_t)row * N + col];
        if (RELU)   v = fmaxf(v, 0.f);
        if (Cout)   Cout[(size_t)row * N + col] = v;
        if (CoutBf) CoutBf[(size_t)row * N + col] = (__bf16)v;
      }
    }
  }
}

// ---------------------------------------------------------------------------
// Small utility kernels
// ---------------------------------------------------------------------------
__global__ void cvt_f32_bf16(const float* __restrict__ in, __bf16* __restrict__ out, long n) {
  long i = (long)blockIdx.x * blockDim.x + threadIdx.x;
  if (i < n) out[i] = (__bf16)in[i];
}

// x: [B,T,FIN] f32 -> xb: [T,B,FIN] bf16 (time-major so per-step rows are contiguous)
__global__ void cvt_x_tmajor(const float* __restrict__ x, __bf16* __restrict__ xb) {
  long i = (long)blockIdx.x * blockDim.x + threadIdx.x;
  const long n = (long)B_ * T_ * FIN_;
  if (i >= n) return;
  int f = (int)(i % FIN_);
  long bt = i / FIN_;
  int t = (int)(bt % T_);
  int b = (int)(bt / T_);
  xb[((long)t * B_ + b) * FIN_ + f] = (__bf16)x[i];
}

__global__ void zero_state(float* h, float* c, __bf16* hb, int n) {
  int i = blockIdx.x * blockDim.x + threadIdx.x;
  if (i < n) { h[i] = 0.f; c[i] = 0.f; hb[i] = (__bf16)0.f; }
}

static __device__ inline float sigm(float x) { return 1.f / (1.f + __expf(-x)); }

// gates: [3][B][4H] f32; h,c: [3][B][H] f32; hbf mirrors h in bf16.
__global__ void lstm_update(const float* __restrict__ gates,
                            float* __restrict__ h, float* __restrict__ c,
                            __bf16* __restrict__ hbf) {
  int i = blockIdx.x * blockDim.x + threadIdx.x;
  if (i >= 3 * B_ * H_) return;
  int cell = i / (B_ * H_);
  int rem  = i - cell * (B_ * H_);
  int b = rem / H_;
  int n = rem - b * H_;
  const float* g = gates + ((size_t)cell * B_ + b) * (4 * H_);
  float ig = sigm(g[n]);
  float fg = sigm(g[H_ + n]);
  float gg = tanhf(g[2 * H_ + n]);
  float og = sigm(g[3 * H_ + n]);
  float cn = fg * c[i] + ig * gg;
  c[i] = cn;
  float hn = og * tanhf(cn);
  h[i] = hn;
  hbf[i] = (__bf16)hn;
}

// One thread per (b, head-class). Writes score buffers inside d_out.
__global__ void heads_kernel(const float* __restrict__ h,  // [3][B][H]
                             const float* __restrict__ Wab, const float* __restrict__ bab,
                             const float* __restrict__ Wac, const float* __restrict__ bac,
                             const float* __restrict__ Wse, const float* __restrict__ bse,
                             float* __restrict__ out, int t,
                             long offAB, long offAC, long offSE) {
  int idx = blockIdx.x * blockDim.x + threadIdx.x;
  const int TOT = 2 + NC_ + 2 * NC_;  // 68
  if (idx >= B_ * TOT) return;
  int b = idx / TOT;
  int ci = idx - b * TOT;
  int cell, cls;
  const float* W; const float* bias; float* dst;
  if (ci < 2)            { cell = 0; cls = ci;           W = Wab; bias = bab; dst = out + offAB + ((long)b * T_ + t) * 2 + cls; }
  else if (ci < 2 + NC_) { cell = 1; cls = ci - 2;       W = Wac; bias = bac; dst = out + offAC + ((long)b * T_ + t) * NC_ + cls; }
  else                   { cell = 2; cls = ci - 2 - NC_; W = Wse; bias = bse; dst = out + offSE + ((long)b * T_ + t) * (2 * NC_) + cls; }
  const float* hh = h + ((size_t)cell * B_ + b) * H_;
  const float* w  = W + (size_t)cls * H_;
  float s = bias[cls];
  for (int k = 0; k < H_; ++k) s = fmaf(hh[k], w[k], s);
  *dst = s;
}

// One thread per batch sample: argmaxes + occurrence logic + one-hot finals.
__global__ void postproc(float* __restrict__ out, long offAB, long offAC, long offSE) {
  int b = threadIdx.x;
  if (b >= B_) return;
  const float* ab = out + offAB + (long)b * T_ * 2;
  const float* ac = out + offAC + (long)b * T_ * NC_;
  const float* se = out + offSE + (long)b * T_ * (2 * NC_);
  int abx[T_], cax[T_], csx[T_];
  unsigned long long occ = 0ull;
  for (int t = 0; t < T_; ++t) {
    // argmax (first max wins, matching jnp.argmax)
    abx[t] = (ab[t * 2 + 1] > ab[t * 2 + 0]) ? 1 : 0;
    int am = 0; float mv = ac[t * NC_];
    for (int c = 1; c < NC_; ++c) { float v = ac[t * NC_ + c]; if (v > mv) { mv = v; am = c; } }
    cax[t] = am;
    int sm = 0; float sv = se[t * 2 * NC_];
    for (int c = 1; c < 2 * NC_; ++c) { float v = se[t * 2 * NC_ + c]; if (v > sv) { sv = v; sm = c; } }
    csx[t] = sm;
    occ |= (1ull << sm);
  }
  for (int t = 0; t < T_; ++t) {
    int cs = csx[t];
    int csm = (cs > NC_ - 1) ? cs - NC_ : cs;
    bool hs = (occ >> csm) & 1ull;
    bool he = (occ >> (csm + NC_)) & 1ull;
    bool is_action = (abx[t] != 0) && ((cax[t] == csm) || (hs && he));
    int fi = is_action ? csm : 0;
    float* row = out + ((long)b * T_ + t) * NC_;
    for (int c = 0; c < NC_; ++c) row[c] = (c == fi) ? 1.f : 0.f;
  }
}

// ---------------------------------------------------------------------------
// Host orchestration
// ---------------------------------------------------------------------------
extern "C" void kernel_launch(void* const* d_in, const int* in_sizes, int n_in,
                              void* d_out, int out_size, void* d_ws, size_t ws_size,
                              hipStream_t stream) {
  (void)in_sizes; (void)n_in; (void)out_size; (void)ws_size;
  const float* x    = (const float*)d_in[0];
  const float* W_fe = (const float*)d_in[1];
  const float* b_fe = (const float*)d_in[2];
  const float* Wih[3] = { (const float*)d_in[3],  (const float*)d_in[9],  (const float*)d_in[15] };
  const float* Whh[3] = { (const float*)d_in[4],  (const float*)d_in[10], (const float*)d_in[16] };
  const float* bih[3] = { (const float*)d_in[5],  (const float*)d_in[11], (const float*)d_in[17] };
  const float* bhh[3] = { (const float*)d_in[6],  (const float*)d_in[12], (const float*)d_in[18] };
  const float* Wc[3]  = { (const float*)d_in[7],  (const float*)d_in[13], (const float*)d_in[19] };
  const float* bc[3]  = { (const float*)d_in[8],  (const float*)d_in[14], (const float*)d_in[20] };
  float* out = (float*)d_out;

  // d_out layout: final [B,T,NC] | actback [B,T,2] | acts [B,T,NC] | startend [B,T,2NC]
  const long offAB = (long)B_ * T_ * NC_;
  const long offAC = offAB + (long)B_ * T_ * 2;
  const long offSE = offAC + (long)B_ * T_ * NC_;

  // Workspace carve-up (256B aligned)
  char* wsb = (char*)d_ws;
  size_t off = 0;
  auto carve = [&](size_t bytes) -> void* {
    off = (off + 255) & ~(size_t)255;
    void* p = wsb + off;
    off += bytes;
    return p;
  };
  __bf16* xb    = (__bf16*)carve((size_t)B_ * T_ * FIN_ * 2);     // [T,B,FIN]
  __bf16* wfeb  = (__bf16*)carve((size_t)FUS_ * FIN_ * 2);
  __bf16* featb = (__bf16*)carve((size_t)B_ * T_ * FUS_ * 2);     // [T*B, FUS]
  __bf16* wihb[3]; for (int p = 0; p < 3; ++p) wihb[p] = (__bf16*)carve((size_t)4 * H_ * FUS_ * 2);
  __bf16* whhb[3]; for (int p = 0; p < 3; ++p) whhb[p] = (__bf16*)carve((size_t)4 * H_ * H_ * 2);
  float*  pre[3];  for (int p = 0; p < 3; ++p) pre[p]  = (float*)carve((size_t)B_ * T_ * 4 * H_ * 4); // [T*B, 4H]
  float*  gates = (float*)carve((size_t)3 * B_ * 4 * H_ * 4);
  float*  hbuf  = (float*)carve((size_t)3 * B_ * H_ * 4);
  float*  cbuf  = (float*)carve((size_t)3 * B_ * H_ * 4);
  __bf16* hbf   = (__bf16*)carve((size_t)3 * B_ * H_ * 2);

  const int TPB = 256;
  auto blocks = [&](long n) { return (unsigned)((n + TPB - 1) / TPB); };

  // 1) Convert weights & inputs to bf16 (one HBM pass, ~negligible vs compute)
  cvt_x_tmajor<<<blocks((long)B_ * T_ * FIN_), TPB, 0, stream>>>(x, xb);
  cvt_f32_bf16<<<blocks((long)FUS_ * FIN_), TPB, 0, stream>>>(W_fe, wfeb, (long)FUS_ * FIN_);
  for (int p = 0; p < 3; ++p) {
    cvt_f32_bf16<<<blocks((long)4 * H_ * FUS_), TPB, 0, stream>>>(Wih[p], wihb[p], (long)4 * H_ * FUS_);
    cvt_f32_bf16<<<blocks((long)4 * H_ * H_),   TPB, 0, stream>>>(Whh[p], whhb[p], (long)4 * H_ * H_);
  }

  // 2) FEAT = ReLU(xb @ W_fe^T + b_fe)   M=4096, N=4096, K=2048  -> bf16
  {
    dim3 grid(FUS_ / 64, (B_ * T_) / 64);
    gemm_wmma_bf16<true><<<grid, 128, 0, stream>>>(
        xb, wfeb, b_fe, nullptr, nullptr, nullptr, featb,
        B_ * T_, FUS_, FIN_);
  }

  // 3) PRE_p = FEAT @ Wih_p^T + bih_p + bhh_p   M=4096, N=4096, K=4096 -> f32
  for (int p = 0; p < 3; ++p) {
    dim3 grid((4 * H_) / 64, (B_ * T_) / 64);
    gemm_wmma_bf16<false><<<grid, 128, 0, stream>>>(
        featb, wihb[p], bih[p], bhh[p], nullptr, pre[p], nullptr,
        B_ * T_, 4 * H_, FUS_);
  }

  // 4) Recurrent chain (the only truly sequential part)
  zero_state<<<blocks(3 * B_ * H_), TPB, 0, stream>>>(hbuf, cbuf, hbf, 3 * B_ * H_);
  for (int t = 0; t < T_; ++t) {
    for (int p = 0; p < 3; ++p) {
      // gates_p = PRE_p[t] + h_p @ Whh_p^T   M=64, N=4096, K=1024
      dim3 grid((4 * H_) / 64, B_ / 64);
      gemm_wmma_bf16<false><<<grid, 128, 0, stream>>>(
          hbf + (size_t)p * B_ * H_, whhb[p], nullptr, nullptr,
          pre[p] + (size_t)t * B_ * 4 * H_,
          gates + (size_t)p * B_ * 4 * H_, nullptr,
          B_, 4 * H_, H_);
    }
    lstm_update<<<blocks(3 * B_ * H_), TPB, 0, stream>>>(gates, hbuf, cbuf, hbf);
    heads_kernel<<<blocks(B_ * 68), TPB, 0, stream>>>(
        hbuf, Wc[0], bc[0], Wc[1], bc[1], Wc[2], bc[2], out, t, offAB, offAC, offSE);
  }

  // 5) Post-processing: argmax / occurrence / one-hot finals
  postproc<<<1, B_, 0, stream>>>(out, offAB, offAC, offSE);
}
